// StiefelAttentionLayer_17987323035911
// MI455X (gfx1250) — matrix operand order repaired and checked by reading, not compiled, not run on hardware
//
#include <hip/hip_runtime.h>
#include <hip/hip_bf16.h>
#include <math.h>

typedef __attribute__((ext_vector_type(16))) __bf16 v16bf;
typedef __attribute__((ext_vector_type(8)))  __bf16 v8bf;
typedef __attribute__((ext_vector_type(8)))  float  v8f;

#define B_   4
#define S_   1024
#define DIM_ 1024
#define H_   16
#define HD_  64

// ---- WMMA fragment loaders (CDNA5 16-bit layouts, CDNA5 ISA) ----
// A (16x32, MxK): lanes 0-15 rows M=0..15 K-base 0, lanes 16-31 same rows K-base 8;
// VGPR 0-3 => K base..base+7 (contiguous), VGPR 4-7 => K base+16..base+23.
__device__ __forceinline__ v16bf frag_a(const __bf16* s, int row0, int k0, int ld) {
  const int lane = threadIdx.x & 31;
  const __bf16* p = s + (row0 + (lane & 15)) * ld + k0 + ((lane >> 4) << 3);
  v8bf lo = *(const v8bf*)p;         // K base .. base+7
  v8bf hi = *(const v8bf*)(p + 16);  // K base+16 .. base+23
  v16bf r;
#pragma unroll
  for (int i = 0; i < 8; ++i) { r[i] = lo[i]; r[i + 8] = hi[i]; }
  return r;
}

// B (32x16, KxN) fed from an N-major (transposed) tile: lane n = lane&15,
// lanes 0-15 hold K=0..7,16..23, lanes 16-31 hold K=8..15,24..31 (same as A).
__device__ __forceinline__ v16bf frag_b(const __bf16* s, int col0, int k0, int ld) {
  const int lane = threadIdx.x & 31;
  const __bf16* p = s + (col0 + (lane & 15)) * ld + k0 + ((lane >> 4) << 3);
  v8bf lo = *(const v8bf*)p;
  v8bf hi = *(const v8bf*)(p + 16);
  v16bf r;
#pragma unroll
  for (int i = 0; i < 8; ++i) { r[i] = lo[i]; r[i + 8] = hi[i]; }
  return r;
}

// ---------------- elementwise fp32 -> bf16 ----------------
__global__ __launch_bounds__(256) void cvt_bf16(const float* __restrict__ src,
                                                __bf16* __restrict__ dst, int n) {
  int i = (blockIdx.x * 256 + threadIdx.x) * 4;
  if (i + 3 < n) {
    float4 f = *(const float4*)(src + i);
    dst[i + 0] = (__bf16)f.x; dst[i + 1] = (__bf16)f.y;
    dst[i + 2] = (__bf16)f.z; dst[i + 3] = (__bf16)f.w;
  }
}

// ---------------- tiled transpose + convert: Wt[n][k] = (bf16)W[k][n] ----------------
__global__ __launch_bounds__(256) void transpose_cvt(const float* __restrict__ W,
                                                     __bf16* __restrict__ Wt,
                                                     int K, int N) {
  __shared__ float tile[32][33];
  const int n0 = blockIdx.x * 32, k0 = blockIdx.y * 32;
  const int tx = threadIdx.x & 31, ty = threadIdx.x >> 5;
#pragma unroll
  for (int i = 0; i < 32; i += 8)
    tile[ty + i][tx] = W[(size_t)(k0 + ty + i) * N + n0 + tx];
  __syncthreads();
#pragma unroll
  for (int i = 0; i < 32; i += 8)
    Wt[(size_t)(n0 + ty + i) * K + k0 + tx] = (__bf16)tile[tx][ty + i];
}

// ---------------- NT GEMM: C[M,N] f32 = A[M,K] bf16 * Bt[N,K]^T bf16 ----------------
// 128x128 block tile, K-step 32, 8 waves; each wave owns a 32x64 region (2x4 16x16 tiles).
__global__ __launch_bounds__(256) void gemm_nt_bf16(const __bf16* __restrict__ A,
                                                    const __bf16* __restrict__ Bt,
                                                    float* __restrict__ C,
                                                    int M, int N, int K) {
  __shared__ __align__(16) __bf16 sA[128 * 32];
  __shared__ __align__(16) __bf16 sB[128 * 32];
  const int tid = threadIdx.x;
  const int lane = tid & 31, wid = tid >> 5;
  const int bm = blockIdx.y * 128, bn = blockIdx.x * 128;
  const int wm = (wid & 3) * 32, wn = (wid >> 2) * 64;
  const int lr = tid >> 1, lc = (tid & 1) * 16;
  v8f acc[2][4] = {};
  for (int k0 = 0; k0 < K; k0 += 32) {
    const __bf16* ga = A + (size_t)(bm + lr) * K + k0 + lc;
    const __bf16* gb = Bt + (size_t)(bn + lr) * K + k0 + lc;
    *(uint4*)&sA[lr * 32 + lc]     = ((const uint4*)ga)[0];
    *(uint4*)&sA[lr * 32 + lc + 8] = ((const uint4*)ga)[1];
    *(uint4*)&sB[lr * 32 + lc]     = ((const uint4*)gb)[0];
    *(uint4*)&sB[lr * 32 + lc + 8] = ((const uint4*)gb)[1];
    if (k0 + 32 < K) {                // pull next K-slice toward L2/L0
      __builtin_prefetch(ga + 32, 0, 0);
      __builtin_prefetch(gb + 32, 0, 0);
    }
    __syncthreads();
#pragma unroll
    for (int mt = 0; mt < 2; ++mt) {
      v16bf af = frag_a(sA, wm + mt * 16, 0, 32);
#pragma unroll
      for (int nt = 0; nt < 4; ++nt) {
        v16bf bfg = frag_b(sB, wn + nt * 16, 0, 32);
        acc[mt][nt] = __builtin_amdgcn_wmma_f32_16x16x32_bf16(
            false, af, false, bfg, (short)0, acc[mt][nt], false, false);
      }
    }
    __syncthreads();
  }
  const int hi8 = (lane >> 4) << 3, cn = lane & 15;
#pragma unroll
  for (int mt = 0; mt < 2; ++mt)
#pragma unroll
    for (int nt = 0; nt < 4; ++nt)
#pragma unroll
      for (int r = 0; r < 8; ++r) {
        int row = bm + wm + mt * 16 + r + hi8;
        int col = bn + wn + nt * 16 + cn;
        C[(size_t)row * N + col] = acc[mt][nt][r];
      }
}

// ---------------- L2-normalize heads, fold kappa into Q, pack bf16, V transposed ----------------
__global__ __launch_bounds__(256) void normalize_qkv(const float* __restrict__ Qf,
                                                     const float* __restrict__ Kf,
                                                     const float* __restrict__ Vf,
                                                     const float* __restrict__ lk,
                                                     __bf16* __restrict__ Qn,
                                                     __bf16* __restrict__ Kn,
                                                     __bf16* __restrict__ Vt) {
  const int seg = blockIdx.x * 8 + (threadIdx.x >> 5);  // one wave per (b,s,h)
  const int lane = threadIdx.x & 31;
  const int h = seg & (H_ - 1);
  const int row = seg >> 4;                 // b*S + s
  const int base = row * (H_ * HD_) + h * HD_;
  const float kappa = __expf(lk[0]);

  float q0 = Qf[base + lane], q1 = Qf[base + 32 + lane];
  float k0 = Kf[base + lane], k1 = Kf[base + 32 + lane];
  float v0 = Vf[base + lane], v1 = Vf[base + 32 + lane];
  float sq = q0 * q0 + q1 * q1, sk = k0 * k0 + k1 * k1, sv = v0 * v0 + v1 * v1;
#pragma unroll
  for (int m = 16; m >= 1; m >>= 1) {
    sq += __shfl_xor(sq, m, 32);
    sk += __shfl_xor(sk, m, 32);
    sv += __shfl_xor(sv, m, 32);
  }
  const float qs = kappa / fmaxf(sqrtf(sq), 1e-12f);
  const float ks = 1.0f  / fmaxf(sqrtf(sk), 1e-12f);
  const float vs = 1.0f  / fmaxf(sqrtf(sv), 1e-12f);
  Qn[base + lane]      = (__bf16)(q0 * qs);
  Qn[base + 32 + lane] = (__bf16)(q1 * qs);
  Kn[base + lane]      = (__bf16)(k0 * ks);
  Kn[base + 32 + lane] = (__bf16)(k1 * ks);
  const int b = row >> 10, s = row & (S_ - 1);
  __bf16* vt = Vt + (size_t)(b * H_ + h) * HD_ * S_ + s;
  vt[(size_t)lane * S_]        = (__bf16)(v0 * vs);
  vt[(size_t)(lane + 32) * S_] = (__bf16)(v1 * vs);
}

// ---------------- flash attention: 64 queries x 64-key tiles, online softmax ----------------
__global__ __launch_bounds__(256) void flash_attn(const __bf16* __restrict__ Qn,
                                                  const __bf16* __restrict__ Kn,
                                                  const __bf16* __restrict__ Vt,
                                                  __bf16* __restrict__ O) {
  __shared__ __align__(16) __bf16 sQ[64 * 64];   // 8 KB
  __shared__ __align__(16) __bf16 sK[64 * 64];   // 8 KB
  __shared__ __align__(16) __bf16 sV[64 * 64];   // 8 KB (V^T tile: [d][k])
  __shared__ __align__(16) float  sS[64 * 64];   // 16 KB logits
  __shared__ __align__(16) __bf16 sP[64 * 64];   // 8 KB probs
  __shared__ float rowM[64], rowL[64], rowA[64];

  const int tid = threadIdx.x, lane = tid & 31, wid = tid >> 5;
  const int bh = blockIdx.x;                 // b*16 + h
  const int b = bh >> 4, h = bh & 15;
  const int q0g = blockIdx.y * 64;

  const int lr = tid >> 2, lc = (tid & 3) * 16;
  {
    const __bf16* g = Qn + (size_t)(b * S_ + q0g + lr) * (H_ * HD_) + h * HD_ + lc;
    *(uint4*)&sQ[lr * 64 + lc]     = ((const uint4*)g)[0];
    *(uint4*)&sQ[lr * 64 + lc + 8] = ((const uint4*)g)[1];
  }
  if (tid < 64) { rowM[tid] = -__builtin_inff(); rowL[tid] = 0.f; }

  // each wave owns 2 of the 16 16x16 tiles of the 64x64 S / O blocks
  const int t0 = wid * 2, t1 = wid * 2 + 1;
  const int mt0 = t0 >> 2, nt0 = t0 & 3;
  const int mt1 = t1 >> 2, nt1 = t1 & 3;
  const int hi8 = (lane >> 4) << 3, cn = lane & 15;
  v8f o0 = {}, o1 = {};

  for (int kt = 0; kt < S_; kt += 64) {
    __syncthreads();
    {
      const __bf16* gk = Kn + (size_t)(b * S_ + kt + lr) * (H_ * HD_) + h * HD_ + lc;
      *(uint4*)&sK[lr * 64 + lc]     = ((const uint4*)gk)[0];
      *(uint4*)&sK[lr * 64 + lc + 8] = ((const uint4*)gk)[1];
      const __bf16* gv = Vt + ((size_t)bh * HD_ + lr) * S_ + kt + lc;
      *(uint4*)&sV[lr * 64 + lc]     = ((const uint4*)gv)[0];
      *(uint4*)&sV[lr * 64 + lc + 8] = ((const uint4*)gv)[1];
    }
    __syncthreads();

    // S = (kappa * Qn) Kn^T
    v8f s0 = {}, s1 = {};
#pragma unroll
    for (int kk = 0; kk < 64; kk += 32) {
      v16bf a0 = frag_a(sQ, mt0 * 16, kk, 64);
      v16bf b0 = frag_b(sK, nt0 * 16, kk, 64);
      s0 = __builtin_amdgcn_wmma_f32_16x16x32_bf16(false, a0, false, b0, (short)0, s0, false, false);
      v16bf a1 = frag_a(sQ, mt1 * 16, kk, 64);
      v16bf b1 = frag_b(sK, nt1 * 16, kk, 64);
      s1 = __builtin_amdgcn_wmma_f32_16x16x32_bf16(false, a1, false, b1, (short)0, s1, false, false);
    }
#pragma unroll
    for (int r = 0; r < 8; ++r) {
      sS[(mt0 * 16 + r + hi8) * 64 + nt0 * 16 + cn] = s0[r];
      sS[(mt1 * 16 + r + hi8) * 64 + nt1 * 16 + cn] = s1[r];
    }
    __syncthreads();

    // online-softmax stats: 4 lanes per row
    {
      const int row = tid >> 2, qp = tid & 3;
      const float* sr = &sS[row * 64 + qp * 16];
      float mx = -__builtin_inff();
#pragma unroll
      for (int i = 0; i < 16; ++i) mx = fmaxf(mx, sr[i]);
      mx = fmaxf(mx, __shfl_xor(mx, 1, 32));
      mx = fmaxf(mx, __shfl_xor(mx, 2, 32));
      if (qp == 0) {
        float mo = rowM[row];
        float mn = fmaxf(mo, mx);
        rowA[row] = __expf(mo - mn);
        rowM[row] = mn;
      }
    }
    __syncthreads();
    {
      const int row = tid >> 2, qp = tid & 3;
      const float m = rowM[row];
      float sum = 0.f;
#pragma unroll
      for (int i = 0; i < 16; ++i) {
        float p = __expf(sS[row * 64 + qp * 16 + i] - m);
        sP[row * 64 + qp * 16 + i] = (__bf16)p;
        sum += p;
      }
      sum += __shfl_xor(sum, 1, 32);
      sum += __shfl_xor(sum, 2, 32);
      if (qp == 0) rowL[row] = rowL[row] * rowA[row] + sum;
    }
    __syncthreads();

    // O = alpha*O + P @ V
#pragma unroll
    for (int r = 0; r < 8; ++r) {
      o0[r] *= rowA[mt0 * 16 + r + hi8];
      o1[r] *= rowA[mt1 * 16 + r + hi8];
    }
#pragma unroll
    for (int kk = 0; kk < 64; kk += 32) {
      v16bf a0 = frag_a(sP, mt0 * 16, kk, 64);
      v16bf b0 = frag_b(sV, nt0 * 16, kk, 64);
      o0 = __builtin_amdgcn_wmma_f32_16x16x32_bf16(false, a0, false, b0, (short)0, o0, false, false);
      v16bf a1 = frag_a(sP, mt1 * 16, kk, 64);
      v16bf b1 = frag_b(sV, nt1 * 16, kk, 64);
      o1 = __builtin_amdgcn_wmma_f32_16x16x32_bf16(false, a1, false, b1, (short)0, o1, false, false);
    }
  }
  __syncthreads();
#pragma unroll
  for (int r = 0; r < 8; ++r) {
    {
      int row = mt0 * 16 + r + hi8;
      int grow = b * S_ + q0g + row;
      O[(size_t)grow * (H_ * HD_) + h * HD_ + nt0 * 16 + cn] = (__bf16)(o0[r] / rowL[row]);
    }
    {
      int row = mt1 * 16 + r + hi8;
      int grow = b * S_ + q0g + row;
      O[(size_t)grow * (H_ * HD_) + h * HD_ + nt1 * 16 + cn] = (__bf16)(o1[r] / rowL[row]);
    }
  }
}

extern "C" void kernel_launch(void* const* d_in, const int* in_sizes, int n_in,
                              void* d_out, int out_size, void* d_ws, size_t ws_size,
                              hipStream_t stream) {
  (void)in_sizes; (void)n_in; (void)out_size; (void)ws_size;
  const float* x  = (const float*)d_in[0];
  const float* Wq = (const float*)d_in[1];
  const float* Wk = (const float*)d_in[2];
  const float* Wv = (const float*)d_in[3];
  const float* Wo = (const float*)d_in[4];
  const float* lk = (const float*)d_in[5];

  char* ws = (char*)d_ws;
  const size_t MB = 1024 * 1024;
  __bf16* xbf = (__bf16*)(ws);             //  8 MB
  __bf16* Wqt = (__bf16*)(ws + 8 * MB);    //  2 MB each
  __bf16* Wkt = (__bf16*)(ws + 10 * MB);
  __bf16* Wvt = (__bf16*)(ws + 12 * MB);
  __bf16* Wot = (__bf16*)(ws + 14 * MB);
  float*  Qf  = (float*)(ws + 16 * MB);    // 16 MB each
  float*  Kf  = (float*)(ws + 32 * MB);
  float*  Vf  = (float*)(ws + 48 * MB);
  __bf16* Qn  = (__bf16*)(ws + 64 * MB);   //  8 MB each
  __bf16* Knb = (__bf16*)(ws + 72 * MB);
  __bf16* Vt  = (__bf16*)(ws + 80 * MB);
  __bf16* Obf = (__bf16*)(ws + 88 * MB);   // -> 96 MB total

  const int M = B_ * S_;    // 4096
  const int N = H_ * HD_;   // 1024
  const int K = DIM_;       // 1024

  cvt_bf16<<<(M * K) / (256 * 4), 256, 0, stream>>>(x, xbf, M * K);
  dim3 tgrid(N / 32, K / 32);
  transpose_cvt<<<tgrid, 256, 0, stream>>>(Wq, Wqt, K, N);
  transpose_cvt<<<tgrid, 256, 0, stream>>>(Wk, Wkt, K, N);
  transpose_cvt<<<tgrid, 256, 0, stream>>>(Wv, Wvt, K, N);
  transpose_cvt<<<tgrid, 256, 0, stream>>>(Wo, Wot, K, N);

  dim3 ggrid(N / 128, M / 128);
  gemm_nt_bf16<<<ggrid, 256, 0, stream>>>(xbf, Wqt, Qf, M, N, K);
  gemm_nt_bf16<<<ggrid, 256, 0, stream>>>(xbf, Wkt, Kf, M, N, K);
  gemm_nt_bf16<<<ggrid, 256, 0, stream>>>(xbf, Wvt, Vf, M, N, K);

  normalize_qkv<<<(B_ * S_ * H_) / 8, 256, 0, stream>>>(Qf, Kf, Vf, lk, Qn, Knb, Vt);

  dim3 fgrid(B_ * H_, S_ / 64);
  flash_attn<<<fgrid, 256, 0, stream>>>(Qn, Knb, Vt, Obf);

  gemm_nt_bf16<<<ggrid, 256, 0, stream>>>(Obf, Wot, (float*)d_out, M, N, K);
}
